// LSTM_79216376808165
// MI455X (gfx1250) — compile-verified, matrix operand order checked by reading, not run on hardware
//
#include <hip/hip_runtime.h>
#include <hip/hip_bf16.h>
#include <math.h>

// ---------------------------------------------------------------------------
// CDNA5 (gfx1250) LSTM scan: bf16 WMMA 16x16x32, f32 accumulate.
// ---------------------------------------------------------------------------

typedef __bf16 bf16;
typedef __attribute__((ext_vector_type(8)))  __bf16 v8bf;
typedef __attribute__((ext_vector_type(16))) __bf16 v16bf;
typedef __attribute__((ext_vector_type(8)))  float  v8f;

constexpr int B  = 128;   // batch
constexpr int T  = 512;   // timesteps
constexpr int NE = 256;   // nemb
constexpr int N  = 1024;  // state
constexpr int NA = 128;   // alphabet
constexpr int KFI = NE + 2 * N;   // 2304
constexpr int KCU = NE + N;       // 1280
constexpr int NKT_FI = KFI / 32;  // 72 k-tiles
constexpr int NKT_CU = KCU / 32;  // 40 k-tiles
constexpr int NKT_N  = N   / 32;  // 32 k-tiles

// ---- fragment helpers -----------------------------------------------------
// A-matrix 16x32 bf16 per ISA: lane m (0..15) holds row m, K runs [0..7],[16..23];
// lanes 16..31 hold row m-16, K runs [8..15],[24..31]. Two 16B loads per frag.
__device__ __forceinline__ v16bf make_frag(const bf16* p) {
  v8bf lo = *(const v8bf*)p;
  v8bf hi = *(const v8bf*)(p + 16);
  v16bf r;
#pragma unroll
  for (int i = 0; i < 8; ++i) { r[i] = lo[i]; r[i + 8] = hi[i]; }
  return r;
}

__device__ __forceinline__ v16bf a_frag(const bf16* tile, int rowStride, int kt, int lane) {
  const int m  = lane & 15;
  const int kb = kt * 32 + ((lane & 16) ? 8 : 0);
  return make_frag(tile + m * rowStride + kb);
}

// B packed layout: frag (ntile, kt) = 32 lanes x 16 contiguous bf16 (1KB tiles).
__device__ __forceinline__ v16bf b_frag(const bf16* packed, int ntile, int nkt, int kt, int lane) {
  return *(const v16bf*)(packed + (((size_t)ntile * nkt + kt) * 32 + lane) * 16);
}

__device__ __forceinline__ v8f wmma_bf16(v16bf a, v16bf b, v8f c) {
  return __builtin_amdgcn_wmma_f32_16x16x32_bf16(false, a, false, b, (short)0, c, false, false);
}

__device__ __forceinline__ float sigmoid_(float v) {
  return 1.0f / (1.0f + __expf(-v));
}
__device__ __forceinline__ float tanh_(float v) {
  float t = __expf(-2.0f * fabsf(v));
  float r = (1.0f - t) / (1.0f + t);
  return copysignf(r, v);
}

// ---- one-time packing -----------------------------------------------------
__global__ __launch_bounds__(256) void cvt_bf16(const float* __restrict__ in,
                                                bf16* __restrict__ out, int total) {
  int i = blockIdx.x * 256 + threadIdx.x;
  if (i < total) out[i] = (bf16)in[i];
}

// W (K x Nn row-major f32) -> packed bf16 WMMA B-fragments.
__global__ __launch_bounds__(256) void pack_weight(const float* __restrict__ W,
                                                   bf16* __restrict__ out,
                                                   int Nn, int nkt, int total) {
  int tid = blockIdx.x * 256 + threadIdx.x;
  if (tid >= total) return;
  int e    = tid & 15;
  int lane = (tid >> 4) & 31;
  int rest = tid >> 9;
  int kt   = rest % nkt;
  int nt   = rest / nkt;
  int k = kt * 32 + ((lane & 16) ? 8 : 0) + ((e < 8) ? e : e + 8);
  int n = nt * 16 + (lane & 15);
  out[tid] = (bf16)W[(size_t)k * Nn + n];
}

// ---- per-step kernel 1: gates + cell update -------------------------------
// grid (8 batch-tiles, 16 n-slices), 128 threads = 4 waves, 16 cols per wave.
__global__ __launch_bounds__(128) void lstm_gates(
    const int* __restrict__ x, const bf16* __restrict__ embbf,
    const bf16* __restrict__ Wfi, const float* __restrict__ bfi,
    const bf16* __restrict__ Wcu, const float* __restrict__ bcu,
    const bf16* __restrict__ h_in, const bf16* __restrict__ cbf_in,
    const float* __restrict__ c32_in, float* __restrict__ c32_out,
    bf16* __restrict__ cbf_out, int t) {
  __shared__ __align__(32) bf16 As[16 * KCU];  // [xt | h] : 40 KB
  const int tid   = threadIdx.x;
  const int lane  = tid & 31;
  const int wave  = tid >> 5;
  const int bbase = blockIdx.x * 16;

  // Stage xt (embedding gather) : 16 x 256
  for (int idx = tid; idx < 16 * (NE / 8); idx += 128) {
    int r = idx >> 5, ch = idx & 31;
    int a = x[(bbase + r) * T + t];
    *(v8bf*)&As[r * KCU + ch * 8] = *(const v8bf*)&embbf[(size_t)a * NE + ch * 8];
  }
  // Stage h : 16 x 1024
  for (int idx = tid; idx < 16 * (N / 8); idx += 128) {
    int r = idx >> 7, ch = idx & 127;
    *(v8bf*)&As[r * KCU + NE + ch * 8] =
        *(const v8bf*)&h_in[(size_t)(bbase + r) * N + ch * 8];
  }
  __syncthreads();

  const int ntile = blockIdx.y * 4 + wave;  // 0..63 in n-space
  v8f accF = {}, accI = {}, accU = {};
  // K < 1280 : one LDS A-frag feeds forget, input, candidate
  for (int kt = 0; kt < NKT_CU; ++kt) {
    v16bf a = a_frag(As, KCU, kt, lane);
    accF = wmma_bf16(a, b_frag(Wfi, ntile,      NKT_FI, kt, lane), accF);
    accI = wmma_bf16(a, b_frag(Wfi, 64 + ntile, NKT_FI, kt, lane), accI);
    accU = wmma_bf16(a, b_frag(Wcu, ntile,      NKT_CU, kt, lane), accU);
  }
  // K = 1280..2303 : A columns come from c (global bf16, L2-hot)
  {
    const int m = lane & 15;
    const bf16* crow = cbf_in + (size_t)(bbase + m) * N + ((lane & 16) ? 8 : 0);
    for (int kt = NKT_CU; kt < NKT_FI; ++kt) {
      v16bf a = make_frag(crow + (kt - NKT_CU) * 32);
      accF = wmma_bf16(a, b_frag(Wfi, ntile,      NKT_FI, kt, lane), accF);
      accI = wmma_bf16(a, b_frag(Wfi, 64 + ntile, NKT_FI, kt, lane), accI);
    }
  }
  // C/D layout: lane fixes column, VGPR index fixes row.
  const int   ncol  = ntile * 16 + (lane & 15);
  const float bF    = bfi[ncol];
  const float bI    = bfi[N + ncol];
  const float bU    = bcu[ncol];
  const int   rbase = bbase + ((lane & 16) ? 8 : 0);
#pragma unroll
  for (int r = 0; r < 8; ++r) {
    size_t off = (size_t)(rbase + r) * N + ncol;
    float f   = sigmoid_(accF[r] + bF);
    float ig  = sigmoid_(accI[r] + bI);
    float u   = tanh_(accU[r] + bU);
    float c2  = f * c32_in[off] + ig * u;
    c32_out[off] = c2;
    cbf_out[off] = (bf16)c2;
  }
}

// ---- per-step kernel 2: hidden projection ---------------------------------
__global__ __launch_bounds__(128) void lstm_hidden(
    const int* __restrict__ x, const bf16* __restrict__ embbf,
    const bf16* __restrict__ Wh, const float* __restrict__ bh,
    const bf16* __restrict__ h_in, const bf16* __restrict__ cbf_new,
    const float* __restrict__ c32_new, bf16* __restrict__ h_out, int t) {
  __shared__ __align__(32) bf16 As[16 * KCU];
  const int tid   = threadIdx.x;
  const int lane  = tid & 31;
  const int wave  = tid >> 5;
  const int bbase = blockIdx.x * 16;

  for (int idx = tid; idx < 16 * (NE / 8); idx += 128) {
    int r = idx >> 5, ch = idx & 31;
    int a = x[(bbase + r) * T + t];
    *(v8bf*)&As[r * KCU + ch * 8] = *(const v8bf*)&embbf[(size_t)a * NE + ch * 8];
  }
  for (int idx = tid; idx < 16 * (N / 8); idx += 128) {
    int r = idx >> 7, ch = idx & 127;
    *(v8bf*)&As[r * KCU + NE + ch * 8] =
        *(const v8bf*)&h_in[(size_t)(bbase + r) * N + ch * 8];
  }
  __syncthreads();

  const int ntile = blockIdx.y * 4 + wave;
  v8f acc = {};
  for (int kt = 0; kt < NKT_CU; ++kt) {
    v16bf a = a_frag(As, KCU, kt, lane);
    acc = wmma_bf16(a, b_frag(Wh, ntile, NKT_FI, kt, lane), acc);
  }
  {
    const int m = lane & 15;
    const bf16* crow = cbf_new + (size_t)(bbase + m) * N + ((lane & 16) ? 8 : 0);
    for (int kt = NKT_CU; kt < NKT_FI; ++kt) {
      v16bf a = make_frag(crow + (kt - NKT_CU) * 32);
      acc = wmma_bf16(a, b_frag(Wh, ntile, NKT_FI, kt, lane), acc);
    }
  }
  const int   ncol  = ntile * 16 + (lane & 15);
  const float bB    = bh[ncol];
  const int   rbase = bbase + ((lane & 16) ? 8 : 0);
#pragma unroll
  for (int r = 0; r < 8; ++r) {
    size_t off = (size_t)(rbase + r) * N + ncol;
    float hg = acc[r] + bB;                 // no sigmoid (faithful to source)
    float h2 = hg * tanh_(c32_new[off]);
    h_out[off] = (bf16)h2;
  }
}

// ---- per-step kernel 3: output projection y[:, t, :] ----------------------
__global__ __launch_bounds__(32) void lstm_out(
    const bf16* __restrict__ h2, const bf16* __restrict__ Wout,
    const float* __restrict__ bout, float* __restrict__ y, int t) {
  const int lane  = threadIdx.x & 31;
  const int bbase = blockIdx.x * 16;
  const int ntile = blockIdx.y;
  const int m     = lane & 15;
  const bf16* arow = h2 + (size_t)(bbase + m) * N + ((lane & 16) ? 8 : 0);
  v8f acc = {};
  for (int kt = 0; kt < NKT_N; ++kt)
    acc = wmma_bf16(make_frag(arow + kt * 32),
                    b_frag(Wout, ntile, NKT_N, kt, lane), acc);
  const int   ncol  = ntile * 16 + (lane & 15);
  const float bb    = bout[ncol];
  const int   rbase = bbase + ((lane & 16) ? 8 : 0);
#pragma unroll
  for (int r = 0; r < 8; ++r)
    y[((size_t)(rbase + r) * T + t) * NA + ncol] = acc[r] + bb;
}

// ---------------------------------------------------------------------------
extern "C" void kernel_launch(void* const* d_in, const int* in_sizes, int n_in,
                              void* d_out, int out_size, void* d_ws, size_t ws_size,
                              hipStream_t stream) {
  const int*   x    = (const int*)  d_in[0];
  const float* emb  = (const float*)d_in[1];
  const float* Wfi  = (const float*)d_in[2];
  const float* bfi  = (const float*)d_in[3];
  const float* Wcu  = (const float*)d_in[4];
  const float* bcu  = (const float*)d_in[5];
  const float* Wh   = (const float*)d_in[6];
  const float* bh   = (const float*)d_in[7];
  const float* Wout = (const float*)d_in[8];
  const float* bout = (const float*)d_in[9];
  float* y = (float*)d_out;

  // workspace carve-out (256B aligned)
  char*  ws  = (char*)d_ws;
  size_t off = 0;
  auto alloc = [&](size_t bytes) -> char* {
    char* p = ws + off;
    off = (off + bytes + 255) & ~(size_t)255;
    return p;
  };
  bf16*  hbuf[2];  float* c32buf[2];  bf16* cbfbuf[2];
  hbuf[0]   = (bf16*) alloc((size_t)B * N * sizeof(bf16));
  hbuf[1]   = (bf16*) alloc((size_t)B * N * sizeof(bf16));
  c32buf[0] = (float*)alloc((size_t)B * N * sizeof(float));
  c32buf[1] = (float*)alloc((size_t)B * N * sizeof(float));
  cbfbuf[0] = (bf16*) alloc((size_t)B * N * sizeof(bf16));
  cbfbuf[1] = (bf16*) alloc((size_t)B * N * sizeof(bf16));
  const size_t stateBytes = off;  // zeroed each launch (h0 = c0 = 0)
  bf16* embbf = (bf16*)alloc((size_t)B  * NE * sizeof(bf16) * 2);  // 128x256 (pad)
  bf16* WfiP  = (bf16*)alloc((size_t)KFI * 2 * N * sizeof(bf16));
  bf16* WcuP  = (bf16*)alloc((size_t)KCU * N     * sizeof(bf16));
  bf16* WhP   = (bf16*)alloc((size_t)KFI * N     * sizeof(bf16));
  bf16* WoutP = (bf16*)alloc((size_t)N   * NA    * sizeof(bf16));
  (void)ws_size; (void)in_sizes; (void)n_in; (void)out_size;

  hipMemsetAsync(d_ws, 0, stateBytes, stream);

  {
    int tot = NA * NE;  // emb is (nalpha=128, nemb=256)
    cvt_bf16<<<(tot + 255) / 256, 256, 0, stream>>>(emb, embbf, tot);
  }
  {
    int tot = KFI * 2 * N;
    pack_weight<<<(tot + 255) / 256, 256, 0, stream>>>(Wfi, WfiP, 2 * N, NKT_FI, tot);
  }
  {
    int tot = KCU * N;
    pack_weight<<<(tot + 255) / 256, 256, 0, stream>>>(Wcu, WcuP, N, NKT_CU, tot);
  }
  {
    int tot = KFI * N;
    pack_weight<<<(tot + 255) / 256, 256, 0, stream>>>(Wh, WhP, N, NKT_FI, tot);
  }
  {
    int tot = N * NA;
    pack_weight<<<(tot + 255) / 256, 256, 0, stream>>>(Wout, WoutP, NA, NKT_N, tot);
  }

  for (int t = 0; t < T; ++t) {
    const int ri = t & 1, wi = ri ^ 1;
    lstm_gates<<<dim3(B / 16, 16), 128, 0, stream>>>(
        x, embbf, WfiP, bfi, WcuP, bcu,
        hbuf[ri], cbfbuf[ri], c32buf[ri], c32buf[wi], cbfbuf[wi], t);
    lstm_hidden<<<dim3(B / 16, 16), 128, 0, stream>>>(
        x, embbf, WhP, bh, hbuf[ri], cbfbuf[wi], c32buf[wi], hbuf[wi], t);
    lstm_out<<<dim3(B / 16, NA / 16), 32, 0, stream>>>(
        hbuf[wi], WoutP, bout, y, t);
  }
}
